// MultiHeadAttention_54460185313542
// MI455X (gfx1250) — compile-verified, matrix operand order checked by reading, not run on hardware
//
#include <hip/hip_runtime.h>
#include <hip/hip_bf16.h>
#include <stdint.h>

// Problem constants (B,S,D,H from the reference)
#define B_  4
#define S_  2048
#define D_  1024
#define H_  16
#define HD_ 64
#define NEG_FILL (-65504.0f)   // fp16 min, matches reference masked_fill

typedef __attribute__((ext_vector_type(16))) _Float16 v16h;
typedef __attribute__((ext_vector_type(8)))  float    v8f;
typedef _Float16 half_t;

union Frag16 { v16h h; uint32_t u[8]; };

// Load a 16x32 f16 fragment in the CDNA5 WMMA A/B register layout.
// Memory is row-major [16 rows][>=32 cols], row r = lane&15, K packed:
// lanes 0-15 hold K 0..7 / 16..23, lanes 16-31 hold K 8..15 / 24..31.
__device__ __forceinline__ v16h load_frag16(const half_t* tile, int ld) {
  const int lane = threadIdx.x & 31;
  const uint32_t* row = reinterpret_cast<const uint32_t*>(tile + (lane & 15) * ld);
  const int hk = (lane >> 4) * 4;            // dword offset for upper half-wave
  Frag16 f;
#pragma unroll
  for (int v = 0; v < 8; ++v) {
    const int kd = ((v & 4) ? 8 : 0) + hk + (v & 3);   // dword index within row
    f.u[v] = row[kd];
  }
  return f.h;
}

__device__ __forceinline__ v8f wmma_f16(v16h a, v16h b, v8f c) {
  return __builtin_amdgcn_wmma_f32_16x16x32_f16(false, a, false, b, (short)0, c, false, false);
}

// ---------------------------------------------------------------- f32 -> f16
__global__ void cvt_f32_f16(const float* __restrict__ in, half_t* __restrict__ out, int n) {
  int i = blockIdx.x * blockDim.x + threadIdx.x;
  const int stride = gridDim.x * blockDim.x;
  for (; i < n; i += stride) out[i] = (half_t)in[i];
}

// ------------------------------------------------------- 128x128 WMMA GEMM
// Y[m,n] = sum_k A[m,k] * W[n,k] + bias[n]   (i.e. A @ W^T + b)
// MODE 0: scatter f16 to Q/K head-major  [b,h,s,hd]
// MODE 1: scatter f16 to V transposed    [b,h,hd,s]
// MODE 2: f32 row-major output [M, D]
template <int MODE>
__global__ __launch_bounds__(256) void gemm128(const half_t* __restrict__ A,
                                               const half_t* __restrict__ W,
                                               const float*  __restrict__ bias,
                                               void* __restrict__ outp) {
  __shared__ __align__(16) half_t xs[128 * 48];
  __shared__ __align__(16) half_t wsh[128 * 48];

  const int m0   = blockIdx.x * 128;
  const int n0   = blockIdx.y * 128;
  const int t    = threadIdx.x;
  const int wave = t >> 5;
  const int lane = t & 31;
  const int wm   = wave & 3;   // 4 row groups of 32
  const int wn   = wave >> 2;  // 2 col groups of 64

  v8f c[2][4] = {};

  const int tr = t >> 1;          // 0..127 : tile row this thread copies
  const int tc = (t & 1) * 16;    // 0 or 16 : first of 16 f16 columns

  for (int kk = 0; kk < D_; kk += 32) {
    if (kk + 32 < D_) {  // prefetch next K slice (global_prefetch_b8)
      __builtin_prefetch(A + (size_t)(m0 + tr) * D_ + kk + 32 + tc, 0, 1);
      __builtin_prefetch(W + (size_t)(n0 + tr) * D_ + kk + 32 + tc, 0, 1);
    }
    const float4* gx = reinterpret_cast<const float4*>(A + (size_t)(m0 + tr) * D_ + kk + tc);
    const float4* gw = reinterpret_cast<const float4*>(W + (size_t)(n0 + tr) * D_ + kk + tc);
    float4* lx = reinterpret_cast<float4*>(xs  + tr * 48 + tc);
    float4* lw = reinterpret_cast<float4*>(wsh + tr * 48 + tc);
    lx[0] = gx[0]; lx[1] = gx[1];
    lw[0] = gw[0]; lw[1] = gw[1];
    __syncthreads();

    v16h a[2], b[4];
#pragma unroll
    for (int i = 0; i < 2; ++i) a[i] = load_frag16(xs  + (wm * 32 + i * 16) * 48, 48);
#pragma unroll
    for (int j = 0; j < 4; ++j) b[j] = load_frag16(wsh + (wn * 64 + j * 16) * 48, 48);

#pragma unroll
    for (int i = 0; i < 2; ++i)
#pragma unroll
      for (int j = 0; j < 4; ++j)
        c[i][j] = wmma_f16(a[i], b[j], c[i][j]);
    __syncthreads();
  }

  // epilogue: C layout -> lane (r, hf, nl): M = r + 8*hf, N = nl
  const int hf = lane >> 4;
  const int nl = lane & 15;
#pragma unroll
  for (int i = 0; i < 2; ++i) {
#pragma unroll
    for (int j = 0; j < 4; ++j) {
#pragma unroll
      for (int r = 0; r < 8; ++r) {
        const int gm = m0 + wm * 32 + i * 16 + r + hf * 8;
        const int gn = n0 + wn * 64 + j * 16 + nl;
        const float v = c[i][j][r] + bias[gn];
        if (MODE == 2) {
          ((float*)outp)[(size_t)gm * D_ + gn] = v;
        } else {
          const int bb = gm >> 11, ss = gm & (S_ - 1);
          const int hh = gn >> 6,  hd = gn & (HD_ - 1);
          if (MODE == 0)
            ((half_t*)outp)[((size_t)(bb * H_ + hh) * S_ + ss) * HD_ + hd] = (half_t)v;
          else
            ((half_t*)outp)[((size_t)(bb * H_ + hh) * HD_ + hd) * S_ + ss] = (half_t)v;
        }
      }
    }
  }
}

// ------------------------------------------------------- flash attention
// One wave = one 16-row q tile of one (b,h). kv streamed in steps of 32.
// Qh,Kh: [b,h,s,hd] f16.  Vt: [b,h,hd,s] f16.  Hh out: [b,s,h*hd] f16.
__global__ __launch_bounds__(256) void flash_attn(const half_t* __restrict__ Qh,
                                                  const half_t* __restrict__ Kh,
                                                  const half_t* __restrict__ Vt,
                                                  const unsigned char* __restrict__ mask,
                                                  half_t* __restrict__ Hh) {
  __shared__ __align__(16) half_t plds[8 * 16 * 32];  // per-wave P buffer
  const int wave = threadIdx.x >> 5;
  const int lane = threadIdx.x & 31;
  const int task = blockIdx.x * 8 + wave;  // 0..8191
  const int qt = task & 127;               // q tile within (b,h)
  const int bh = task >> 7;                // 0..63
  const int b  = bh >> 4;
  const int h  = bh & 15;
  const int q0 = qt * 16;

  const half_t* Qb = Qh + (size_t)bh * S_ * HD_;
  const half_t* Kb = Kh + (size_t)bh * S_ * HD_;
  const half_t* Vb = Vt + (size_t)bh * HD_ * S_;
  const unsigned char* Mb = mask + (size_t)b * S_ * S_;

  // resident Q fragments (hd 0..31 and 32..63)
  const v16h aq0 = load_frag16(Qb + (size_t)q0 * HD_,      HD_);
  const v16h aq1 = load_frag16(Qb + (size_t)q0 * HD_ + 32, HD_);

  v8f o[4] = {};
  float mrun[8], lrun[8];
#pragma unroll
  for (int r = 0; r < 8; ++r) { mrun[r] = -3.0e38f; lrun[r] = 0.0f; }

  half_t* P = plds + wave * (16 * 32);
  const int hf = lane >> 4;
  const int nl = lane & 15;

  for (int kv = 0; kv < S_; kv += 32) {
    // S = Q * K^T  (two 16-col kv tiles, two hd k-steps each)
    v8f s0 = {}, s1 = {};
    s0 = wmma_f16(aq0, load_frag16(Kb + (size_t)kv * HD_,             HD_), s0);
    s0 = wmma_f16(aq1, load_frag16(Kb + (size_t)kv * HD_ + 32,        HD_), s0);
    s1 = wmma_f16(aq0, load_frag16(Kb + (size_t)(kv + 16) * HD_,      HD_), s1);
    s1 = wmma_f16(aq1, load_frag16(Kb + (size_t)(kv + 16) * HD_ + 32, HD_), s1);

    // scale + boolean mask -> fp16 min
#pragma unroll
    for (int r = 0; r < 8; ++r) {
      const int qrow = q0 + r + hf * 8;
      const unsigned char mb0 = Mb[(size_t)qrow * S_ + kv + nl];
      const unsigned char mb1 = Mb[(size_t)qrow * S_ + kv + 16 + nl];
      s0[r] = mb0 ? s0[r] * 0.125f : NEG_FILL;
      s1[r] = mb1 ? s1[r] * 0.125f : NEG_FILL;
    }

    // online softmax per row (row = r + 8*hf, spread over 16 lanes of same half)
#pragma unroll
    for (int r = 0; r < 8; ++r) {
      float mx = fmaxf(s0[r], s1[r]);
      mx = fmaxf(mx, __shfl_xor(mx, 1));
      mx = fmaxf(mx, __shfl_xor(mx, 2));
      mx = fmaxf(mx, __shfl_xor(mx, 4));
      mx = fmaxf(mx, __shfl_xor(mx, 8));
      const float mnew  = fmaxf(mrun[r], mx);
      const float alpha = __expf(mrun[r] - mnew);
      const float p0 = __expf(s0[r] - mnew);
      const float p1 = __expf(s1[r] - mnew);
      float rs = p0 + p1;
      rs += __shfl_xor(rs, 1);
      rs += __shfl_xor(rs, 2);
      rs += __shfl_xor(rs, 4);
      rs += __shfl_xor(rs, 8);
      lrun[r] = lrun[r] * alpha + rs;
      mrun[r] = mnew;
#pragma unroll
      for (int j = 0; j < 4; ++j) o[j][r] *= alpha;
      // stash P (f16) in wave-private LDS for C-layout -> A-layout transpose
      P[(r + hf * 8) * 32 + nl]      = (half_t)p0;
      P[(r + hf * 8) * 32 + 16 + nl] = (half_t)p1;
    }
    asm volatile("s_wait_dscnt 0" ::: "memory");

    // O += P * V   (k = 32 kv rows, 4 hd tiles of 16)
    const v16h ap = load_frag16(P, 32);
#pragma unroll
    for (int j = 0; j < 4; ++j) {
      const v16h bv = load_frag16(Vb + (size_t)(j * 16) * S_ + kv, S_);
      o[j] = wmma_f16(ap, bv, o[j]);
    }
  }

  // normalize and write head output back to [b, s, h*64+hd] f16
#pragma unroll
  for (int r = 0; r < 8; ++r) {
    const int qrow = q0 + r + hf * 8;
    const float inv = 1.0f / lrun[r];
#pragma unroll
    for (int j = 0; j < 4; ++j)
      Hh[(size_t)(b * S_ + qrow) * D_ + h * HD_ + j * 16 + nl] = (half_t)(o[j][r] * inv);
  }
}

// ----------------------------------------------------------------- launch
extern "C" void kernel_launch(void* const* d_in, const int* in_sizes, int n_in,
                              void* d_out, int out_size, void* d_ws, size_t ws_size,
                              hipStream_t stream) {
  const float* x  = (const float*)d_in[0];
  const unsigned char* mask = (const unsigned char*)d_in[1];
  const float* Wq = (const float*)d_in[2];
  const float* bq = (const float*)d_in[3];
  const float* Wk = (const float*)d_in[4];
  const float* bk = (const float*)d_in[5];
  const float* Wv = (const float*)d_in[6];
  const float* bv = (const float*)d_in[7];
  const float* Wf = (const float*)d_in[8];
  const float* bf = (const float*)d_in[9];
  float* out = (float*)d_out;

  const size_t M  = (size_t)B_ * S_;          // 8192
  const size_t XE = M * D_;                   // x / H elements
  const size_t WE = (size_t)D_ * D_;          // weight elements

  char* p = (char*)d_ws;
  half_t* xh  = (half_t*)p; p += XE * sizeof(half_t);
  half_t* wqh = (half_t*)p; p += WE * sizeof(half_t);
  half_t* wkh = (half_t*)p; p += WE * sizeof(half_t);
  half_t* wvh = (half_t*)p; p += WE * sizeof(half_t);
  half_t* wfh = (half_t*)p; p += WE * sizeof(half_t);
  half_t* Qh  = (half_t*)p; p += XE * sizeof(half_t);
  half_t* Kh  = (half_t*)p; p += XE * sizeof(half_t);
  half_t* Vt  = (half_t*)p; p += XE * sizeof(half_t);
  half_t* Hh  = (half_t*)p; p += XE * sizeof(half_t);

  cvt_f32_f16<<<4096, 256, 0, stream>>>(x,  xh,  (int)XE);
  cvt_f32_f16<<<1024, 256, 0, stream>>>(Wq, wqh, (int)WE);
  cvt_f32_f16<<<1024, 256, 0, stream>>>(Wk, wkh, (int)WE);
  cvt_f32_f16<<<1024, 256, 0, stream>>>(Wv, wvh, (int)WE);
  cvt_f32_f16<<<1024, 256, 0, stream>>>(Wf, wfh, (int)WE);

  dim3 g((unsigned)(M / 128), (unsigned)(D_ / 128));
  gemm128<0><<<g, 256, 0, stream>>>(xh, wqh, bq, (void*)Qh);
  gemm128<0><<<g, 256, 0, stream>>>(xh, wkh, bk, (void*)Kh);
  gemm128<1><<<g, 256, 0, stream>>>(xh, wvh, bv, (void*)Vt);

  flash_attn<<<(unsigned)(B_ * H_ * (S_ / 16) / 8), 256, 0, stream>>>(Qh, Kh, Vt, mask, Hh);

  gemm128<2><<<g, 256, 0, stream>>>(Hh, wfh, bf, (void*)out);
}
